// LocationSensitiveAttention_62594853371962
// MI455X (gfx1250) — matrix operand order repaired and checked by reading, not compiled, not run on hardware
//
#include <hip/hip_runtime.h>
#include <hip/hip_bf16.h>
#include <math.h>

typedef __bf16 bf16;
typedef __attribute__((ext_vector_type(16))) __bf16 v16bf;
typedef __attribute__((ext_vector_type(8)))  float  v8f;

#define B_   64
#define T_   2048
#define ENC_ 512
#define DEC_ 1024
#define ATT_ 128
#define NF_  32
#define KS_  31
#define PAD_ 15

// ---------------- prep kernels ----------------

// dec_proj[b,a] = decoder_state[b,:] . W_decoder[a,:]   (64*128 outputs, K=1024)
__global__ void __launch_bounds__(256) k_dec_proj(const float* __restrict__ dec,
                                                  const float* __restrict__ Wd,
                                                  float* __restrict__ out) {
    int idx = blockIdx.x * 256 + threadIdx.x;      // 8192
    int b = idx >> 7, a = idx & 127;
    const float* x = dec + b * DEC_;
    const float* w = Wd  + a * DEC_;
    float s = 0.f;
    for (int d = 0; d < DEC_; ++d) s += x[d] * w[d];
    out[idx] = s;
}

// W_encoder fp32 -> bf16, same [A=128][E=512] row-major layout
__global__ void __launch_bounds__(256) k_cvt_wenc(const float* __restrict__ w,
                                                  bf16* __restrict__ o) {
    int i = blockIdx.x * 256 + threadIdx.x;        // 65536
    o[i] = (bf16)w[i];
}

// W_eff[a,k] = sum_f W_location[a,f] * conv_w[f,0,k], k<31; k==31 -> 0  (bf16 [128][32])
__global__ void __launch_bounds__(256) k_loc_eff(const float* __restrict__ Wl,
                                                 const float* __restrict__ cw,
                                                 bf16* __restrict__ o) {
    int idx = blockIdx.x * 256 + threadIdx.x;      // 4096
    int a = idx >> 5, k = idx & 31;
    float s = 0.f;
    if (k < KS_) {
        for (int f = 0; f < NF_; ++f) s += Wl[a * NF_ + f] * cw[f * KS_ + k];
    }
    o[a * 32 + k] = (bf16)s;
}

// ---------------- main energy kernel (WMMA) ----------------
// grid (B, T/128), block 256 = 8 waves; wave -> 16 rows of T x 128 cols of A
__device__ __forceinline__ void cvt8_into(v16bf& d, int base, float4 a, float4 b) {
    d[base + 0] = (bf16)a.x; d[base + 1] = (bf16)a.y;
    d[base + 2] = (bf16)a.z; d[base + 3] = (bf16)a.w;
    d[base + 4] = (bf16)b.x; d[base + 5] = (bf16)b.y;
    d[base + 6] = (bf16)b.z; d[base + 7] = (bf16)b.w;
}

__global__ void __launch_bounds__(256) k_energy(const float* __restrict__ enc,
                                                const float* __restrict__ prev,
                                                const float* __restrict__ dec_proj,
                                                const float* __restrict__ vvec,
                                                const bf16*  __restrict__ WencB,
                                                const bf16*  __restrict__ WlocB,
                                                float* __restrict__ energy) {
    const int b     = blockIdx.x;
    const int wave  = threadIdx.x >> 5;
    const int lane  = threadIdx.x & 31;
    const int t0    = blockIdx.y * 128 + wave * 16;
    const int mrow  = lane & 15;           // A-matrix row within tile for this lane
    const int khalf = lane >> 4;           // lane<16 -> K 0..15 ; lane>=16 -> K 16..31
    const int row   = t0 + mrow;
    const int ncol  = lane & 15;           // B/C column within N-tile

    v8f acc[8];
#pragma unroll
    for (int nt = 0; nt < 8; ++nt) {
#pragma unroll
        for (int j = 0; j < 8; ++j) acc[nt][j] = 0.f;
    }

    // ---- location term: A = sliding window of prev (31 taps, tap 31 zero in W_eff)
    v16bf aloc;
    {
        const float* pb = prev + (size_t)b * T_;
#pragma unroll
        for (int e = 0; e < 8; ++e) {
            int j1 = khalf * 16 + e;                // K 16h+0..7
            int i1 = row - PAD_ + j1;
            aloc[e]     = (bf16)((i1 >= 0 && i1 < T_) ? pb[i1] : 0.f);
            int j2 = khalf * 16 + 8 + e;            // K 16h+8..15
            int i2 = row - PAD_ + j2;
            aloc[8 + e] = (bf16)((i2 >= 0 && i2 < T_) ? pb[i2] : 0.f);
        }
    }
#pragma unroll
    for (int nt = 0; nt < 8; ++nt) {
        v16bf bfrag = *reinterpret_cast<const v16bf*>(WlocB + (nt * 16 + ncol) * 32 + khalf * 16);
        acc[nt] = __builtin_amdgcn_wmma_f32_16x16x32_bf16(
            false, aloc, false, bfrag, (short)0, acc[nt], false, false);
    }

    // ---- encoder projection: [16 rows x 512] x [512 x 128], K in 16 steps of 32
    const float* arow = enc + ((size_t)(b * T_ + row)) * ENC_;
#pragma unroll
    for (int kk = 0; kk < 16; ++kk) {
        v16bf afrag;
        const float4* p0 = reinterpret_cast<const float4*>(arow + kk * 32 + khalf * 16);
        const float4* p1 = reinterpret_cast<const float4*>(arow + kk * 32 + khalf * 16 + 8);
        cvt8_into(afrag, 0, p0[0], p0[1]);
        cvt8_into(afrag, 8, p1[0], p1[1]);
#pragma unroll
        for (int nt = 0; nt < 8; ++nt) {
            v16bf bfrag = *reinterpret_cast<const v16bf*>(
                WencB + (nt * 16 + ncol) * ENC_ + kk * 32 + khalf * 16);
            acc[nt] = __builtin_amdgcn_wmma_f32_16x16x32_bf16(
                false, afrag, false, bfrag, (short)0, acc[nt], false, false);
        }
    }

    // ---- epilogue: + dec_proj, tanh, dot v, reduce over A (8 ntiles x 16 lanes)
    float decv[8], vv[8];
#pragma unroll
    for (int nt = 0; nt < 8; ++nt) {
        decv[nt] = dec_proj[b * ATT_ + nt * 16 + ncol];
        vv[nt]   = vvec[nt * 16 + ncol];
    }
#pragma unroll
    for (int j = 0; j < 8; ++j) {
        float part = 0.f;
#pragma unroll
        for (int nt = 0; nt < 8; ++nt)
            part += vv[nt] * tanhf(acc[nt][j] + decv[nt]);
        // sum across the 16 lanes of each half (lanes 0-15: M=j, lanes 16-31: M=j+8)
#pragma unroll
        for (int m = 1; m < 16; m <<= 1) part += __shfl_xor(part, m, 32);
        if (lane == 0)  energy[b * T_ + t0 + j]     = part;
        if (lane == 16) energy[b * T_ + t0 + 8 + j] = part;
    }
}

// ---------------- softmax over T per batch ----------------
__global__ void __launch_bounds__(256) k_softmax(const float* __restrict__ energy,
                                                 const int* __restrict__ mask,
                                                 float* __restrict__ out) {
    __shared__ float red[256];
    const int b = blockIdx.x, tid = threadIdx.x;
    float* ctx  = out;                  // [64*512], zero for the context pass
    float* wout = out + B_ * ENC_;      // [64*2048]
    for (int e = tid; e < ENC_; e += 256) ctx[b * ENC_ + e] = 0.f;

    float ev[8];
    float lmax = -3.4e38f;
#pragma unroll
    for (int i = 0; i < 8; ++i) {
        int t = tid + i * 256;
        float e = energy[b * T_ + t];
        if (mask[b * T_ + t] == 0) e = -1e9f;
        ev[i] = e;
        lmax = fmaxf(lmax, e);
    }
    red[tid] = lmax; __syncthreads();
    for (int s = 128; s > 0; s >>= 1) {
        if (tid < s) red[tid] = fmaxf(red[tid], red[tid + s]);
        __syncthreads();
    }
    float m = red[0]; __syncthreads();

    float lsum = 0.f;
#pragma unroll
    for (int i = 0; i < 8; ++i) { ev[i] = __expf(ev[i] - m); lsum += ev[i]; }
    red[tid] = lsum; __syncthreads();
    for (int s = 128; s > 0; s >>= 1) {
        if (tid < s) red[tid] += red[tid + s];
        __syncthreads();
    }
    float inv = 1.f / red[0];
#pragma unroll
    for (int i = 0; i < 8; ++i) wout[b * T_ + tid + i * 256] = ev[i] * inv;
}

// ---------------- context = sum_t w[b,t] * enc[b,t,:] ----------------
// grid (B, 2 e-chunks, 8 t-chunks), 256 threads; fp32 atomics (8 partials/output)
__global__ void __launch_bounds__(256) k_context(const float* __restrict__ enc,
                                                 const float* __restrict__ w,
                                                 float* __restrict__ ctx) {
    const int b = blockIdx.x;
    const int e = blockIdx.y * 256 + threadIdx.x;
    const int tbase = blockIdx.z * 256;
    const float* wb = w + (size_t)b * T_ + tbase;
    const float* eb = enc + ((size_t)b * T_ + tbase) * ENC_ + e;
    float s = 0.f;
    for (int t = 0; t < 256; ++t) s += wb[t] * eb[(size_t)t * ENC_];
    atomicAdd(&ctx[b * ENC_ + e], s);
}

// ---------------- launcher ----------------
extern "C" void kernel_launch(void* const* d_in, const int* in_sizes, int n_in,
                              void* d_out, int out_size, void* d_ws, size_t ws_size,
                              hipStream_t stream) {
    const float* dec    = (const float*)d_in[0];   // [64,1024]
    const float* enc    = (const float*)d_in[1];   // [64,2048,512]
    const float* prev   = (const float*)d_in[2];   // [64,2048]
    const int*   mask   = (const int*)  d_in[3];   // [64,2048]
    const float* conv_w = (const float*)d_in[4];   // [32,1,31]
    const float* WencF  = (const float*)d_in[5];   // [128,512]
    const float* Wdec   = (const float*)d_in[6];   // [128,1024]
    const float* WlocF  = (const float*)d_in[7];   // [128,32]
    const float* v      = (const float*)d_in[8];   // [1,128]
    float* out = (float*)d_out;                    // [64*512 context | 64*2048 weights]

    char* ws = (char*)d_ws;
    float* dec_proj = (float*)(ws);                                  // 32 KB
    float* energy   = (float*)(ws + 32768);                          // 512 KB
    bf16*  WencB    = (bf16*) (ws + 32768 + 524288);                 // 128 KB
    bf16*  WlocB    = (bf16*) (ws + 32768 + 524288 + 131072);        // 8 KB

    k_dec_proj<<<32, 256, 0, stream>>>(dec, Wdec, dec_proj);
    k_cvt_wenc<<<256, 256, 0, stream>>>(WencF, WencB);
    k_loc_eff<<<16, 256, 0, stream>>>(WlocF, conv_w, WlocB);

    dim3 ge(B_, T_ / 128);
    k_energy<<<ge, 256, 0, stream>>>(enc, prev, dec_proj, v, WencB, WlocB, energy);

    k_softmax<<<B_, 256, 0, stream>>>(energy, mask, out);

    dim3 gc(B_, ENC_ / 256, T_ / 256);
    k_context<<<gc, 256, 0, stream>>>(enc, out + B_ * ENC_, out);
}
